// UltimateTwoTowerMoERecModel_38345468019295
// MI455X (gfx1250) — matrix operand order repaired and checked by reading, not compiled, not run on hardware
//
#include <hip/hip_runtime.h>
#include <hip/hip_bf16.h>
#include <math.h>

// ---------------------------------------------------------------------------
// MI455X (gfx1250) implementation. Compute-bound model (~77 GFLOP, weights fit
// in L2) -> all large GEMMs go through v_wmma_f32_16x16x32_bf16 (bf16 inputs,
// fp32 accumulate). Tiles staged in LDS as packed bf16 in fragment-linear
// order so each lane builds its WMMA fragments with ds_load_b128s. Each wave
// computes a 16x64 output slab (A fragment reused across 4 wmma issues);
// epilogue variants are compile-time template instantiations.
// ---------------------------------------------------------------------------

typedef __bf16 bf16_t;
typedef bf16_t   v16bf  __attribute__((ext_vector_type(16)));
typedef float    v8f    __attribute__((ext_vector_type(8)));
typedef float    f32x4  __attribute__((ext_vector_type(4)));
typedef unsigned u32x2  __attribute__((ext_vector_type(2)));
typedef unsigned u32x4  __attribute__((ext_vector_type(4)));

#define NHEADS 8
#define DMODEL 256
#define HDIM   32

#define GF_GELU  1
#define GF_ACC   2
#define GF_SCALE 4

__device__ __forceinline__ float gelu_exact(float x) {
  return 0.5f * x * (1.f + erff(x * 0.70710678118654752440f));
}

__device__ __forceinline__ unsigned pk2(float a, float b) {
  union { bf16_t h[2]; unsigned u; } p;
  p.h[0] = (bf16_t)a; p.h[1] = (bf16_t)b;
  return p.u;
}

__device__ __forceinline__ float wave_sum(float v) {
  for (int o = 16; o; o >>= 1) v += __shfl_xor(v, o, 32);
  return v;
}
__device__ __forceinline__ float wave_max(float v) {
  for (int o = 16; o; o >>= 1) v = fmaxf(v, __shfl_xor(v, o, 32));
  return v;
}

// ------------------------- embedding gather + RoPE -------------------------
__global__ void k_gather_rope(const int* __restrict__ hist,
                              const float* __restrict__ emb,
                              float* __restrict__ X, int T, int S) {
  int idx = blockIdx.x * blockDim.x + threadIdx.x;
  if (idx >= T * DMODEL) return;
  int t = idx >> 8, d = idx & 255;
  int s = t % S;
  const float* row = emb + (size_t)hist[t] * DMODEL;
  float x0 = row[d];
  float x1 = row[(d + 1) & 255];                       // roll(x,-1) over D
  float inv = __powf(10000.f, -(float)(2 * (d & 127)) / 256.f);
  float ang = (float)s * inv;
  X[idx] = x0 * __cosf(ang) + x1 * __sinf(ang);
}

// ------------------------------ WMMA GEMM ----------------------------------
// C[M,N] = epilogue(A[M,K] @ W[N,K]^T + bias[N])
// 4 waves / block; each wave owns a 16(M) x 64(N) slab (4 wmma per k-step,
// A fragment reused). Block covers 16 x 256 of C. LDS tiles are packed bf16
// (2 per dword) in row-major [row][K] order => fragments are contiguous:
//   A lane frag: row ml, K = half*8..+7 and 16+half*8..+7   (2x ds_load_b128)
//   B lane frag: row n,  K = half*8..+7 and 16+half*8..+7   (2x ds_load_b128)
template <int FLAGS>
__global__ __launch_bounds__(128)
void k_gemm_wmma(const float* __restrict__ A, const float* __restrict__ W,
                 const float* __restrict__ bias, float* __restrict__ C,
                 int M, int N, int K,
                 const float* __restrict__ rs, int rs_stride, int rs_off) {
  __shared__ unsigned Asb[16][20];       // 16 rows x 32 bf16 (16 u32 + 4 pad)
  __shared__ unsigned Bsb[4][64][20];    // per wave: 64 n-rows x 32 bf16
  const int tid  = threadIdx.x;
  const int w    = tid >> 5, lane = tid & 31;
  const int m0   = blockIdx.x * 16;
  const int n0w  = (blockIdx.y * 4 + w) * 64;
  const bool act = (n0w < N);
  const int half = lane >> 4;
  const int ml   = lane & 15;

  v8f acc[4] = {};
  const int ksteps = K >> 5;
  for (int kk = 0; kk < ksteps; ++kk) {
    const int k0 = kk << 5;
    if (kk + 1 < ksteps) {               // prefetch next K tile (global_prefetch_b8)
      __builtin_prefetch(&A[(size_t)(m0 + ml) * K + k0 + 32], 0, 1);
      if (act) __builtin_prefetch(&W[(size_t)(n0w + lane) * K + k0 + 32], 0, 1);
    }
    // cooperative A tile: each thread converts 4 consecutive fp32 -> bf16x4
    {
      int i = tid << 2;                  // 0..508
      int r = i >> 5, c = i & 31;
      f32x4 f = *(const f32x4*)&A[(size_t)(m0 + r) * K + k0 + c];
      u32x2 pr; pr[0] = pk2(f[0], f[1]); pr[1] = pk2(f[2], f[3]);
      *(u32x2*)&Asb[r][c >> 1] = pr;
    }
    // per-wave B tile: lane converts 2 full n-rows (32 fp32 each)
    if (act) {
      #pragma unroll
      for (int rr = 0; rr < 64; rr += 32) {
        const float* wr = &W[(size_t)(n0w + rr + lane) * K + k0];
        #pragma unroll
        for (int c = 0; c < 32; c += 4) {
          f32x4 f = *(const f32x4*)&wr[c];
          u32x2 pr; pr[0] = pk2(f[0], f[1]); pr[1] = pk2(f[2], f[3]);
          *(u32x2*)&Bsb[w][rr + lane][c >> 1] = pr;
        }
      }
    }
    __syncthreads();
    if (act) {
      union FragU { u32x4 q[2]; v16bf v; } fa, fb;
      fa.q[0] = *(const u32x4*)&Asb[ml][half * 4];
      fa.q[1] = *(const u32x4*)&Asb[ml][8 + half * 4];
      #pragma unroll
      for (int s = 0; s < 4; ++s) {
        fb.q[0] = *(const u32x4*)&Bsb[w][s * 16 + ml][half * 4];
        fb.q[1] = *(const u32x4*)&Bsb[w][s * 16 + ml][8 + half * 4];
        acc[s] = __builtin_amdgcn_wmma_f32_16x16x32_bf16(
            false, fa.v, false, fb.v, (short)0, acc[s], false, false);
      }
    }
    __syncthreads();
  }
  if (act) {
    #pragma unroll
    for (int s = 0; s < 4; ++s) {
      int n = n0w + s * 16 + ml;
      float bn = bias[n];
      #pragma unroll
      for (int i = 0; i < 8; ++i) {
        int m = m0 + (half << 3) + i;                // C layout (ISA 7.12.2)
        float v = acc[s][i] + bn;
        if constexpr (FLAGS & GF_GELU)  v = gelu_exact(v);
        if constexpr (FLAGS & GF_SCALE) v *= rs[(size_t)m * rs_stride + rs_off];
        size_t off = (size_t)m * N + n;
        if constexpr (FLAGS & GF_ACC) C[off] += v; else C[off] = v;
      }
    }
  }
}

// ------------------------------- attention ---------------------------------
// One block per (batch, head). K tile staged in LDS, wave-per-query-row.
__global__ __launch_bounds__(256)
void k_attn(const float* __restrict__ Qb, const float* __restrict__ Kb,
            const float* __restrict__ Vb, float* __restrict__ Ob,
            int nq, int ns, int qstride, int kvstride, int ostride,
            float scale) {
  __shared__ float Ks[200 * 32];
  __shared__ float Ps[8][200];
  const int tid = threadIdx.x;
  const int bh = blockIdx.x;
  const int b = bh / NHEADS, h = bh % NHEADS;
  const int koff = h * HDIM;
  for (int i = tid; i < ns * 32; i += 256) {
    int j = i >> 5, d = i & 31;
    Ks[i] = Kb[(size_t)(b * ns + j) * kvstride + koff + d];
  }
  __syncthreads();
  const int w = tid >> 5, lane = tid & 31;
  const int jblocks = (ns + 31) >> 5;
  for (int r = w; r < nq; r += 8) {           // uniform trip count across waves
    float q = Qb[(size_t)(b * nq + r) * qstride + koff + lane];
    for (int jj = 0; jj < jblocks; ++jj) {
      int j = (jj << 5) + lane;
      float s = 0.f;
      #pragma unroll
      for (int d = 0; d < 32; ++d) s += __shfl(q, d, 32) * Ks[j * 32 + d];
      if (j < ns) Ps[w][j] = s * scale;
    }
    float mx = -3.4e38f;
    for (int jj = 0; jj < jblocks; ++jj) {
      int j = (jj << 5) + lane;
      if (j < ns) mx = fmaxf(mx, Ps[w][j]);
    }
    mx = wave_max(mx);
    float sum = 0.f;
    for (int jj = 0; jj < jblocks; ++jj) {
      int j = (jj << 5) + lane;
      if (j < ns) { float e = __expf(Ps[w][j] - mx); Ps[w][j] = e; sum += e; }
    }
    sum = wave_sum(sum);
    float invs = 1.f / sum;
    __syncthreads();
    float o = 0.f;
    for (int j = 0; j < ns; ++j)
      o += Ps[w][j] * Vb[(size_t)(b * ns + j) * kvstride + koff + lane];
    Ob[(size_t)(b * nq + r) * ostride + koff + lane] = o * invs;
    __syncthreads();
  }
}

// ---------------------- residual + LayerNorm (D=256) -----------------------
__global__ __launch_bounds__(256)
void k_addln(const float* __restrict__ X, const float* __restrict__ R,
             const float* __restrict__ s, const float* __restrict__ bp,
             float* __restrict__ O) {
  __shared__ float red[256];
  const int t = blockIdx.x, d = threadIdx.x;
  float v = X[(size_t)t * DMODEL + d];
  if (R) v += R[(size_t)t * DMODEL + d];
  red[d] = v; __syncthreads();
  for (int o = 128; o; o >>= 1) { if (d < o) red[d] += red[d + o]; __syncthreads(); }
  float mean = red[0] * (1.f / 256.f);
  __syncthreads();
  float dv = v - mean;
  red[d] = dv * dv; __syncthreads();
  for (int o = 128; o; o >>= 1) { if (d < o) red[d] += red[d + o]; __syncthreads(); }
  float var = red[0] * (1.f / 256.f);
  O[(size_t)t * DMODEL + d] = dv * rsqrtf(var + 1e-5f) * s[d] + bp[d];
}

// ---- MoE gate: logits -> softmax -> top-2 combine weights + aux accum -----
__global__ __launch_bounds__(256)
void k_gate(const float* __restrict__ X, const float* __restrict__ gw,
            const float* __restrict__ gb, float* __restrict__ cw,
            float* __restrict__ auxacc, int T, int E) {
  __shared__ float pa[8];
  const int tid = threadIdx.x;
  if (tid < 8) pa[tid] = 0.f;
  __syncthreads();
  const int t = blockIdx.x * 256 + tid;
  if (t < T) {
    float l[8];
    const float* x = X + (size_t)t * DMODEL;
    for (int e = 0; e < E; ++e) {
      float s = gb[e];
      const float* we = gw + (size_t)e * DMODEL;
      for (int d = 0; d < DMODEL; ++d) s += x[d] * we[d];
      l[e] = s;
    }
    float mx = l[0];
    for (int e = 1; e < E; ++e) mx = fmaxf(mx, l[e]);
    float sum = 0.f;
    for (int e = 0; e < E; ++e) { l[e] = __expf(l[e] - mx); sum += l[e]; }
    float inv = 1.f / sum;
    for (int e = 0; e < E; ++e) { l[e] *= inv; atomicAdd(&pa[e], l[e]); }
    int i1 = 0;
    for (int e = 1; e < E; ++e) if (l[e] > l[i1]) i1 = e;   // first-max tie rule
    int i2 = (i1 == 0) ? 1 : 0;
    for (int e = 0; e < E; ++e) if (e != i1 && l[e] > l[i2]) i2 = e;
    float nrm = 1.f / (l[i1] + l[i2]);
    for (int e = 0; e < E; ++e) cw[(size_t)t * E + e] = 0.f;
    cw[(size_t)t * E + i1] = l[i1] * nrm;
    cw[(size_t)t * E + i2] = l[i2] * nrm;
  }
  __syncthreads();
  if (tid < E) atomicAdd(&auxacc[tid], pa[tid]);
}

// ------------------------------ small kernels ------------------------------
__global__ void k_item_gather(const int* __restrict__ cand,
                              const int* __restrict__ cat,
                              const float* __restrict__ iemb,
                              const float* __restrict__ cemb,
                              float* __restrict__ IE, int TI) {
  int idx = blockIdx.x * blockDim.x + threadIdx.x;
  if (idx >= TI * DMODEL) return;
  int i = idx >> 8, d = idx & 255;
  IE[idx] = iemb[(size_t)cand[i] * DMODEL + d] + cemb[(size_t)cat[i] * DMODEL + d];
}

__global__ void k_concat(const float* __restrict__ A, const float* __restrict__ B2,
                         float* __restrict__ F, int TI) {
  int idx = blockIdx.x * blockDim.x + threadIdx.x;
  if (idx >= TI * 512) return;
  int i = idx >> 9, c = idx & 511;
  F[idx] = (c < 256) ? A[(size_t)i * 256 + c] : B2[(size_t)i * 256 + (c - 256)];
}

__global__ void k_final(const float* __restrict__ H, const float* __restrict__ w2,
                        const float* __restrict__ b2, float* __restrict__ out, int TI) {
  int t = blockIdx.x * blockDim.x + threadIdx.x;
  if (t >= TI) return;
  float s = b2[0];
  const float* h = H + (size_t)t * 256;
  for (int d = 0; d < 256; ++d) s += h[d] * w2[d];
  out[t] = s;
}

__global__ void k_aux(const float* __restrict__ acc, float* __restrict__ out) {
  const int Ts[3] = {12800, 12800, 1024};
  const int Es[3] = {4, 4, 8};
  float aux = 0.f;
  for (int r = 0; r < 3; ++r) {
    float s = 0.f;
    for (int e = 0; e < Es[r]; ++e) {
      float d = acc[r * 8 + e] / (float)Ts[r] - 0.125f;   // 1/NE_CFG = 1/8
      s += d * d;
    }
    aux += s / (float)Es[r];
  }
  out[1024] = aux;
}

// ------------------------------- host side ---------------------------------
static inline void gemm(hipStream_t st, const float* A, const float* W,
                        const float* b, float* C, int M, int N, int K,
                        int flags = 0, const float* rs = nullptr,
                        int rss = 0, int rso = 0) {
  dim3 g(M / 16, (N + 255) / 256);
  if (flags == GF_GELU)
    k_gemm_wmma<GF_GELU><<<g, 128, 0, st>>>(A, W, b, C, M, N, K, rs, rss, rso);
  else if (flags == (GF_ACC | GF_SCALE))
    k_gemm_wmma<GF_ACC | GF_SCALE><<<g, 128, 0, st>>>(A, W, b, C, M, N, K, rs, rss, rso);
  else
    k_gemm_wmma<0><<<g, 128, 0, st>>>(A, W, b, C, M, N, K, rs, rss, rso);
}

extern "C" void kernel_launch(void* const* d_in, const int* in_sizes, int n_in,
                              void* d_out, int out_size, void* d_ws, size_t ws_size,
                              hipStream_t stream) {
  (void)in_sizes; (void)n_in; (void)out_size; (void)ws_size;
  const int*   hist     = (const int*)  d_in[0];
  const int*   cand     = (const int*)  d_in[1];
  const int*   catid    = (const int*)  d_in[2];
  const float* item_emb = (const float*)d_in[3];
  const float* cat_emb  = (const float*)d_in[4];
  const float* ub_in_w  = (const float*)d_in[5];
  const float* ub_in_b  = (const float*)d_in[6];
  const float* ub_out_w = (const float*)d_in[7];
  const float* ub_out_b = (const float*)d_in[8];
  const float* ub_n1_s  = (const float*)d_in[9];
  const float* ub_n1_b  = (const float*)d_in[10];
  const float* ub_n2_s  = (const float*)d_in[11];
  const float* ub_n2_b  = (const float*)d_in[12];
  const float* ub_gw    = (const float*)d_in[13];
  const float* ub_gb    = (const float*)d_in[14];
  const float* ub_w1    = (const float*)d_in[15];
  const float* ub_b1    = (const float*)d_in[16];
  const float* ub_w2    = (const float*)d_in[17];
  const float* ub_b2    = (const float*)d_in[18];
  const float* un_s     = (const float*)d_in[19];
  const float* un_b     = (const float*)d_in[20];
  const float* im_gw    = (const float*)d_in[21];
  const float* im_gb    = (const float*)d_in[22];
  const float* im_w1    = (const float*)d_in[23];
  const float* im_b1    = (const float*)d_in[24];
  const float* im_w2    = (const float*)d_in[25];
  const float* im_b2    = (const float*)d_in[26];
  const float* in_s     = (const float*)d_in[27];
  const float* in_b     = (const float*)d_in[28];
  const float* ca_in_w  = (const float*)d_in[29];
  const float* ca_in_b  = (const float*)d_in[30];
  const float* ca_out_w = (const float*)d_in[31];
  const float* ca_out_b = (const float*)d_in[32];
  const float* fm_w1    = (const float*)d_in[33];
  const float* fm_b1    = (const float*)d_in[34];
  const float* fm_w2    = (const float*)d_in[35];
  const float* fm_b2    = (const float*)d_in[36];
  float* out = (float*)d_out;

  const int T = 64 * 200, TI = 64 * 16, S = 200;
  const float scale = 0.17677669529663687f;   // 1/sqrt(32)

  // workspace layout (fp32 elements)
  float* wsf  = (float*)d_ws;
  float* X    = wsf;                      // [12800,256]
  float* QKV  = X    + (size_t)T * 256;   // [12800,768] (reused: cross K/V 512)
  float* TMP  = QKV  + (size_t)T * 768;   // [12800,256]
  float* Hb   = TMP  + (size_t)T * 256;   // [12800,512]
  float* CW   = Hb   + (size_t)T * 512;   // [12800,8]
  float* IE   = CW   + (size_t)T * 8;     // [1024,256]
  float* ITEM = IE   + (size_t)TI * 256;  // [1024,256]
  float* QI   = ITEM + (size_t)TI * 256;  // [1024,256]
  float* TAW  = QI   + (size_t)TI * 256;  // [1024,256]
  float* FUS  = TAW  + (size_t)TI * 256;  // [1024,512]
  float* FH   = FUS  + (size_t)TI * 512;  // [1024,256]
  float* AUX  = FH   + (size_t)TI * 256;  // [3*8] softmax-prob accumulators

  hipMemsetAsync(AUX, 0, 32 * sizeof(float), stream);

  // ---- user tower: embed + rope ----
  k_gather_rope<<<(T * 256 + 255) / 256, 256, 0, stream>>>(hist, item_emb, X, T, S);

  for (int l = 0; l < 2; ++l) {
    // self-attention
    gemm(stream, X, ub_in_w + (size_t)l * 768 * 256, ub_in_b + l * 768,
         QKV, T, 768, 256);
    k_attn<<<64 * NHEADS, 256, 0, stream>>>(QKV, QKV + 256, QKV + 512, TMP,
                                            S, S, 768, 768, 256, scale);
    gemm(stream, TMP, ub_out_w + (size_t)l * 256 * 256, ub_out_b + l * 256,
         Hb, T, 256, 256);
    k_addln<<<T, 256, 0, stream>>>(X, Hb, ub_n1_s + l * 256, ub_n1_b + l * 256, X);
    // MoE (EU=4, top-2): dense experts, combine folded into FFN2 epilogue
    k_gate<<<(T + 255) / 256, 256, 0, stream>>>(
        X, ub_gw + (size_t)l * 4 * 256, ub_gb + l * 4, CW, AUX + l * 8, T, 4);
    hipMemsetAsync(TMP, 0, (size_t)T * 256 * sizeof(float), stream);
    for (int e = 0; e < 4; ++e) {
      const size_t xe = (size_t)(l * 4 + e);
      gemm(stream, X, ub_w1 + xe * 512 * 256, ub_b1 + xe * 512,
           Hb, T, 512, 256, GF_GELU);
      gemm(stream, Hb, ub_w2 + xe * 256 * 512, ub_b2 + xe * 256,
           TMP, T, 256, 512, GF_ACC | GF_SCALE, CW, 4, e);
    }
    k_addln<<<T, 256, 0, stream>>>(X, TMP, ub_n2_s + l * 256, ub_n2_b + l * 256, X);
  }
  // hist = LN(x)
  k_addln<<<T, 256, 0, stream>>>(X, nullptr, un_s, un_b, X);

  // ---- item tower ----
  k_item_gather<<<(TI * 256 + 255) / 256, 256, 0, stream>>>(
      cand, catid, item_emb, cat_emb, IE, TI);
  k_gate<<<(TI + 255) / 256, 256, 0, stream>>>(IE, im_gw, im_gb, CW, AUX + 16, TI, 8);
  hipMemsetAsync(ITEM, 0, (size_t)TI * 256 * sizeof(float), stream);
  for (int e = 0; e < 8; ++e) {
    gemm(stream, IE, im_w1 + (size_t)e * 512 * 256, im_b1 + (size_t)e * 512,
         Hb, TI, 512, 256, GF_GELU);
    gemm(stream, Hb, im_w2 + (size_t)e * 256 * 512, im_b2 + (size_t)e * 256,
         ITEM, TI, 256, 512, GF_ACC | GF_SCALE, CW, 8, e);
  }
  k_addln<<<TI, 256, 0, stream>>>(ITEM, nullptr, in_s, in_b, ITEM);

  // ---- cross attention: item queries over history ----
  gemm(stream, ITEM, ca_in_w, ca_in_b, QI, TI, 256, 256);                 // Q
  gemm(stream, X, ca_in_w + 256 * 256, ca_in_b + 256, QKV, T, 512, 256); // K,V
  k_attn<<<64 * NHEADS, 256, 0, stream>>>(QI, QKV, QKV + 256, TMP,
                                          16, S, 256, 512, 256, scale);
  gemm(stream, TMP, ca_out_w, ca_out_b, TAW, TI, 256, 256);

  // ---- fusion MLP + logits + aux ----
  k_concat<<<(TI * 512 + 255) / 256, 256, 0, stream>>>(TAW, ITEM, FUS, TI);
  gemm(stream, FUS, fm_w1, fm_b1, FH, TI, 256, 512, GF_GELU);
  k_final<<<(TI + 255) / 256, 256, 0, stream>>>(FH, fm_w2, fm_b2, out, TI);
  k_aux<<<1, 1, 0, stream>>>(AUX, out);
}
